// AdaptiveMixtureLayer_45105746542642
// MI455X (gfx1250) — compile-verified, matrix operand order checked by reading, not run on hardware
//
#include <hip/hip_runtime.h>
#include <hip/hip_bf16.h>

// ---------------------------------------------------------------------------
// AdaptiveMixtureLayer for MI455X (gfx1250, wave32, WMMA)
// N=2048 atoms, D=256, H=8 heads, DK=32.  All matmuls via
// v_wmma_f32_16x16x32_bf16.  Flash-style fused softmax + adjacency blend.
// ---------------------------------------------------------------------------

typedef __bf16 bf16_t;
typedef __attribute__((ext_vector_type(16))) __bf16 v16bf;
typedef __attribute__((ext_vector_type(8)))  __bf16 v8bf;
typedef __attribute__((ext_vector_type(8)))  float  v8f;
typedef __attribute__((ext_vector_type(4)))  float  v4f;

constexpr int  NA   = 2048;   // atoms
constexpr int  DM   = 256;    // d_model
constexpr int  DK   = 32;     // head dim
constexpr float QSCALE = 0.17677669529663687f; // 1/sqrt(32)

// D = A(16x32 bf16) x B(32x16 bf16) + C  (f32 accum)
__device__ __forceinline__ v8f wmma_bf16(v16bf a, v16bf b, v8f c) {
  return __builtin_amdgcn_wmma_f32_16x16x32_bf16(false, a, false, b,
                                                 (short)0, c, false, false);
}

// Load a 16-element bf16 lane-fragment: two contiguous 8-elem (16B) chunks.
__device__ __forceinline__ v16bf load_frag_bf16(const bf16_t* p0,
                                                const bf16_t* p1) {
  v8bf lo = *reinterpret_cast<const v8bf*>(p0);
  v8bf hi = *reinterpret_cast<const v8bf*>(p1);
  v16bf r;
#pragma unroll
  for (int i = 0; i < 8; ++i) { r[i] = lo[i]; r[8 + i] = hi[i]; }
  return r;
}

// Load 2x8 contiguous fp32 and convert to a bf16 fragment.
__device__ __forceinline__ v16bf cvt_frag_f32(const float* p0,
                                              const float* p1) {
  v4f a0 = *reinterpret_cast<const v4f*>(p0);
  v4f a1 = *reinterpret_cast<const v4f*>(p0 + 4);
  v4f b0 = *reinterpret_cast<const v4f*>(p1);
  v4f b1 = *reinterpret_cast<const v4f*>(p1 + 4);
  v16bf r;
#pragma unroll
  for (int i = 0; i < 4; ++i) {
    r[i]      = (bf16_t)a0[i];
    r[4 + i]  = (bf16_t)a1[i];
    r[8 + i]  = (bf16_t)b0[i];
    r[12 + i] = (bf16_t)b1[i];
  }
  return r;
}

// ---------------------------------------------------------------------------
// 1) inverse row sums of adjacency: inv[r] = 1/(sum_c adj[r,c] + 1e-6)
// ---------------------------------------------------------------------------
__global__ void adj_rowsum_inv_kernel(const float* __restrict__ adj,
                                      float* __restrict__ inv) {
  __shared__ float red[256];
  const int row = blockIdx.x;
  const float* p = adj + (size_t)row * NA;
  float s = 0.f;
  for (int c = threadIdx.x; c < NA; c += 256) s += p[c];
  red[threadIdx.x] = s;
  __syncthreads();
  for (int off = 128; off > 0; off >>= 1) {
    if (threadIdx.x < off) red[threadIdx.x] += red[threadIdx.x + off];
    __syncthreads();
  }
  if (threadIdx.x == 0) inv[row] = 1.0f / (red[0] + 1e-6f);
}

// ---------------------------------------------------------------------------
// 2) fused Q/K/V projection:  out^T tile = W_rows x X_rows^T  (WMMA)
//    mat 0: Q = (x@Wq^T+bq) * 1/sqrt(DK)  -> bf16 [N, D] row-major
//    mat 1: K =  x@Wk^T+bk               -> bf16 [N, D] row-major
//    mat 2: V =  x@Wv^T+bv               -> bf16 [D, N] (transposed!)
// wave tile: 16 output features (j) x 16 atoms (n); 6144 waves total.
// ---------------------------------------------------------------------------
__global__ void qkv_proj_kernel(const float* __restrict__ q_in,
                                const float* __restrict__ k_in,
                                const float* __restrict__ v_in,
                                const float* __restrict__ Wq, const float* __restrict__ bq,
                                const float* __restrict__ Wk, const float* __restrict__ bk,
                                const float* __restrict__ Wv, const float* __restrict__ bv,
                                bf16_t* __restrict__ Qbf,
                                bf16_t* __restrict__ Kbf,
                                bf16_t* __restrict__ Vt) {
  const int wid  = blockIdx.x * (blockDim.x >> 5) + (threadIdx.x >> 5);
  const int lane = threadIdx.x & 31;
  const int half = lane >> 4, lr = lane & 15;

  const int mat = wid / (16 * 128);
  const int rem = wid % (16 * 128);
  const int jt  = rem & 15;   // 16 feature tiles
  const int nt  = rem >> 4;   // 128 atom tiles

  const float* X = (mat == 0) ? q_in : (mat == 1) ? k_in : v_in;
  const float* W = (mat == 0) ? Wq   : (mat == 1) ? Wk   : Wv;
  const float* b = (mat == 0) ? bq   : (mat == 1) ? bk   : bv;

  const int jrow = jt * 16 + lr;   // A fragment: M = feature row
  const int nrow = nt * 16 + lr;   // B fragment: N = atom (column)

  v8f acc = {};
#pragma unroll
  for (int d0 = 0; d0 < DM; d0 += 32) {
    const int c0 = d0 + half * 8, c1 = d0 + 16 + half * 8;
    v16bf A = cvt_frag_f32(W + (size_t)jrow * DM + c0, W + (size_t)jrow * DM + c1);
    v16bf B = cvt_frag_f32(X + (size_t)nrow * DM + c0, X + (size_t)nrow * DM + c1);
    acc = wmma_bf16(A, B, acc);
  }

  const float scale = (mat == 0) ? QSCALE : 1.0f;
  const int jbase = jt * 16 + half * 8;   // C frag rows: j = jbase + v
#pragma unroll
  for (int v = 0; v < 8; ++v) acc[v] = (acc[v] + b[jbase + v]) * scale;

  if (mat < 2) {
    bf16_t* dst = (mat == 0) ? Qbf : Kbf;
    v8bf o;
#pragma unroll
    for (int v = 0; v < 8; ++v) o[v] = (bf16_t)acc[v];
    *reinterpret_cast<v8bf*>(dst + (size_t)nrow * DM + jbase) = o;  // [n, j]
  } else {
#pragma unroll
    for (int v = 0; v < 8; ++v)
      Vt[(size_t)(jbase + v) * NA + nrow] = (bf16_t)acc[v];         // [j, n]
  }
}

// ---------------------------------------------------------------------------
// 3) fused attention + adjacency blend (flash-style, transposed fragments).
//    One wave per (head, 16-query-row block).  Per 32-key chunk:
//      S^T  = K_chunk @ Q^T                 (2 WMMA)
//      online softmax stats per query row   (per-lane + shfl_xor 16)
//      acc^T += V^T_chunk @ P^T             (2 WMMA)  -- P^T packed directly
//      adj^T += V^T_chunk @ Padj^T          (2 WMMA)  -- from the C fragment
//    out = 0.5*acc/l + 0.5*adj, stored as bf16 x[n, d].
// ---------------------------------------------------------------------------
__global__ void fused_attn_adj_kernel(const float* __restrict__ mask,
                                      const float* __restrict__ adj,
                                      const float* __restrict__ invsum,
                                      const bf16_t* __restrict__ Qbf,
                                      const bf16_t* __restrict__ Kbf,
                                      const bf16_t* __restrict__ Vt,
                                      bf16_t* __restrict__ Xbf) {
  const int wid  = blockIdx.x * (blockDim.x >> 5) + (threadIdx.x >> 5);
  const int lane = threadIdx.x & 31;
  const int half = lane >> 4, lr = lane & 15;

  const int h    = wid >> 7;       // head
  const int nt   = wid & 127;      // query-row tile
  const int qrow = nt * 16 + lr;
  const int hd   = h * DK;

  // Q fragment (B operand, column = query row), Q pre-scaled by 1/sqrt(DK)
  const bf16_t* qp = Qbf + (size_t)qrow * DM + hd;
  const v16bf Qfrag = load_frag_bf16(qp + half * 8, qp + 16 + half * 8);

  const float* mrow = mask + (size_t)qrow * NA;
  const float* arow = adj  + (size_t)qrow * NA;
  const float invs  = invsum[qrow];

  v8f acc0 = {}, acc1 = {}, aadj0 = {}, aadj1 = {};
  float m = -1e30f, l = 0.f;

  for (int kb = 0; kb < NA; kb += 32) {
    // --- S^T = K @ Q^T : A = key rows, 2 tiles of 16 keys ---
    const bf16_t* kp0 = Kbf + (size_t)(kb + lr) * DM + hd;
    const bf16_t* kp1 = Kbf + (size_t)(kb + 16 + lr) * DM + hd;
    v16bf Ak0 = load_frag_bf16(kp0 + half * 8, kp0 + 16 + half * 8);
    v16bf Ak1 = load_frag_bf16(kp1 + half * 8, kp1 + 16 + half * 8);
    v8f z = {};
    v8f s0 = wmma_bf16(Ak0, Qfrag, z);   // keys kb + half*8 + v
    v8f s1 = wmma_bf16(Ak1, Qfrag, z);   // keys kb + 16 + half*8 + v

    // --- additive mask (coalesced along keys for fixed qrow) ---
    {
      const float* mp0 = mrow + kb + half * 8;
      const float* mp1 = mrow + kb + 16 + half * 8;
      v4f m00 = *reinterpret_cast<const v4f*>(mp0);
      v4f m01 = *reinterpret_cast<const v4f*>(mp0 + 4);
      v4f m10 = *reinterpret_cast<const v4f*>(mp1);
      v4f m11 = *reinterpret_cast<const v4f*>(mp1 + 4);
#pragma unroll
      for (int v = 0; v < 4; ++v) {
        s0[v] += m00[v]; s0[4 + v] += m01[v];
        s1[v] += m10[v]; s1[4 + v] += m11[v];
      }
    }

    // --- online softmax: stats per query row (= per lane, halves merged) ---
    float tmax = s0[0];
#pragma unroll
    for (int v = 1; v < 8; ++v) tmax = fmaxf(tmax, s0[v]);
#pragma unroll
    for (int v = 0; v < 8; ++v) tmax = fmaxf(tmax, s1[v]);
    tmax = fmaxf(tmax, __shfl_xor(tmax, 16, 32));
    const float mn   = fmaxf(m, tmax);
    const float corr = __expf(m - mn);
    float p0[8], p1[8], lsum = 0.f;
#pragma unroll
    for (int v = 0; v < 8; ++v) {
      p0[v] = __expf(s0[v] - mn);
      p1[v] = __expf(s1[v] - mn);
      lsum += p0[v] + p1[v];
    }
    lsum += __shfl_xor(lsum, 16, 32);
    l = l * corr + lsum;
    m = mn;
#pragma unroll
    for (int v = 0; v < 8; ++v) { acc0[v] *= corr; acc1[v] *= corr; }

    // --- pack P^T straight into the bf16 B-operand layout ---
    v16bf Bp;
#pragma unroll
    for (int v = 0; v < 8; ++v) { Bp[v] = (bf16_t)p0[v]; Bp[8 + v] = (bf16_t)p1[v]; }

    // --- V^T A-fragments: dims hd..hd+15 and hd+16..hd+31, keys kb..kb+31 ---
    const bf16_t* vp0 = Vt + (size_t)(hd + lr) * NA + kb;
    const bf16_t* vp1 = Vt + (size_t)(hd + 16 + lr) * NA + kb;
    v16bf Av0 = load_frag_bf16(vp0 + half * 8, vp0 + 16 + half * 8);
    v16bf Av1 = load_frag_bf16(vp1 + half * 8, vp1 + 16 + half * 8);
    acc0 = wmma_bf16(Av0, Bp, acc0);
    acc1 = wmma_bf16(Av1, Bp, acc1);

    // --- adjacency B-operand: row-normalized adjacency, same key chunk ---
    const float* ap0 = arow + kb + half * 8;
    const float* ap1 = arow + kb + 16 + half * 8;
    v4f a00 = *reinterpret_cast<const v4f*>(ap0);
    v4f a01 = *reinterpret_cast<const v4f*>(ap0 + 4);
    v4f a10 = *reinterpret_cast<const v4f*>(ap1);
    v4f a11 = *reinterpret_cast<const v4f*>(ap1 + 4);
    v16bf Ba;
#pragma unroll
    for (int v = 0; v < 4; ++v) {
      Ba[v]      = (bf16_t)(a00[v] * invs);
      Ba[4 + v]  = (bf16_t)(a01[v] * invs);
      Ba[8 + v]  = (bf16_t)(a10[v] * invs);
      Ba[12 + v] = (bf16_t)(a11[v] * invs);
    }
    aadj0 = wmma_bf16(Av0, Ba, aadj0);
    aadj1 = wmma_bf16(Av1, Ba, aadj1);
  }

  // out = 0.5 * attn/l + 0.5 * adj   (l is per-lane = per query row)
  const float rl = 0.5f / l;
  v8bf o0, o1;
#pragma unroll
  for (int v = 0; v < 8; ++v) {
    o0[v] = (bf16_t)(acc0[v] * rl + 0.5f * aadj0[v]);
    o1[v] = (bf16_t)(acc1[v] * rl + 0.5f * aadj1[v]);
  }
  bf16_t* xp = Xbf + (size_t)qrow * DM + hd;
  *reinterpret_cast<v8bf*>(xp + half * 8)      = o0;  // dims hd+half*8+v
  *reinterpret_cast<v8bf*>(xp + 16 + half * 8) = o1;  // dims hd+16+half*8+v
}

// ---------------------------------------------------------------------------
// 4) output projection: out[n,j] = x(bf16) @ Wo^T + bo   (fp32 result)
// ---------------------------------------------------------------------------
__global__ void out_proj_kernel(const bf16_t* __restrict__ Xbf,
                                const float* __restrict__ Wo,
                                const float* __restrict__ bo,
                                float* __restrict__ out) {
  const int wid  = blockIdx.x * (blockDim.x >> 5) + (threadIdx.x >> 5);
  const int lane = threadIdx.x & 31;
  const int half = lane >> 4, lr = lane & 15;
  const int jt = wid & 15;
  const int nt = wid >> 4;
  const int jrow = jt * 16 + lr;
  const int nrow = nt * 16 + lr;

  v8f acc = {};
#pragma unroll
  for (int d0 = 0; d0 < DM; d0 += 32) {
    const int c0 = d0 + half * 8, c1 = d0 + 16 + half * 8;
    v16bf A = cvt_frag_f32(Wo + (size_t)jrow * DM + c0, Wo + (size_t)jrow * DM + c1);
    const bf16_t* xp = Xbf + (size_t)nrow * DM;
    v16bf B = load_frag_bf16(xp + c0, xp + c1);
    acc = wmma_bf16(A, B, acc);
  }
  const int jbase = jt * 16 + half * 8;
  float* op = out + (size_t)nrow * DM + jbase;
  v4f lo, hi;
#pragma unroll
  for (int v = 0; v < 4; ++v) {
    lo[v] = acc[v] + bo[jbase + v];
    hi[v] = acc[4 + v] + bo[jbase + 4 + v];
  }
  *reinterpret_cast<v4f*>(op)     = lo;
  *reinterpret_cast<v4f*>(op + 4) = hi;
}

// ---------------------------------------------------------------------------
extern "C" void kernel_launch(void* const* d_in, const int* in_sizes, int n_in,
                              void* d_out, int out_size, void* d_ws, size_t ws_size,
                              hipStream_t stream) {
  (void)in_sizes; (void)n_in; (void)out_size; (void)ws_size;
  const float* query = (const float*)d_in[0];
  const float* key_  = (const float*)d_in[1];
  const float* value = (const float*)d_in[2];
  const float* mask  = (const float*)d_in[3];
  const float* adj   = (const float*)d_in[4];
  const float* Wq = (const float*)d_in[5];   const float* bq = (const float*)d_in[6];
  const float* Wk = (const float*)d_in[7];   const float* bk = (const float*)d_in[8];
  const float* Wv = (const float*)d_in[9];   const float* bv = (const float*)d_in[10];
  const float* Wo = (const float*)d_in[11];  const float* bo = (const float*)d_in[12];
  float* out = (float*)d_out;

  char* ws = (char*)d_ws;
  bf16_t* Qbf = (bf16_t*)(ws + (size_t)0 * (1u << 20));  // [N, D] bf16, pre-scaled
  bf16_t* Kbf = (bf16_t*)(ws + (size_t)1 * (1u << 20));  // [N, D] bf16
  bf16_t* Vt  = (bf16_t*)(ws + (size_t)2 * (1u << 20));  // [D, N] bf16 (transposed)
  bf16_t* Xbf = (bf16_t*)(ws + (size_t)3 * (1u << 20));  // [N, D] bf16
  float* invsum = (float*)(ws + (size_t)4 * (1u << 20)); // [N] fp32

  adj_rowsum_inv_kernel<<<NA, 256, 0, stream>>>(adj, invsum);
  qkv_proj_kernel<<<(3 * 16 * 128) / 8, 256, 0, stream>>>(
      query, key_, value, Wq, bq, Wk, bk, Wv, bv, Qbf, Kbf, Vt);
  fused_attn_adj_kernel<<<(8 * 128) / 8, 256, 0, stream>>>(
      mask, adj, invsum, Qbf, Kbf, Vt, Xbf);
  out_proj_kernel<<<(16 * 128) / 8, 256, 0, stream>>>(Xbf, Wo, bo, out);
}
